// MultiHeadSelfAttention_21844203668092
// MI455X (gfx1250) — matrix operand order repaired and checked by reading, not compiled, not run on hardware
//
#include <hip/hip_runtime.h>
#include <hip/hip_bf16.h>
#include <stdint.h>

// ---------------------------------------------------------------------------
// MI455X (gfx1250) multi-head self-attention, bf16 WMMA path + async TDM
// double buffering.  B=4, N=2048, DIM=1024, H=16, DH=64.  wave32.
// ---------------------------------------------------------------------------

#define BSZ   4
#define SEQ   2048
#define DIM   1024
#define HEADS 16
#define DHEAD 64
#define QKVC  (3 * HEADS * DHEAD)   // 3072

typedef __bf16    bf16x16 __attribute__((ext_vector_type(16)));
typedef __bf16    bf16x8  __attribute__((ext_vector_type(8)));
typedef float     f32x8   __attribute__((ext_vector_type(8)));
typedef uint32_t  u32x4   __attribute__((ext_vector_type(4)));
typedef uint32_t  u32x8   __attribute__((ext_vector_type(8)));

// D = A(16x32 bf16) * B(32x16 bf16) + C(16x16 f32)
static __device__ inline f32x8 wmma_bf16(bf16x16 a, bf16x16 b, f32x8 c) {
  return __builtin_amdgcn_wmma_f32_16x16x32_bf16(
      false, a, false, b, (short)0, c, false, false);
}

// A-fragment (16x32, 16-bit): lane = half*16 + M; element e holds
// K = (e&7) + 8*half + 16*(e>>3).  Per lane: two contiguous 16B chunks.
static __device__ inline bf16x16 load_a_frag(const __bf16* base, int ld) {
  int lane = threadIdx.x & 31;
  int m    = lane & 15;
  int half = lane >> 4;
  bf16x16 a;
#pragma unroll
  for (int e = 0; e < 16; ++e) {
    int k = (e & 7) + 8 * half + ((e >> 3) << 4);
    a[e] = base[m * ld + k];
  }
  return a;
}

// B-fragment (32x16) from a TRANSPOSED LDS image: logical B(k,n) stored at
// base[n*ld + k]; element e holds K = (e&7) + 8*half + 16*(e>>3) -> two 16B runs.
static __device__ inline bf16x16 load_bT_frag(const __bf16* base, int ld) {
  int lane = threadIdx.x & 31;
  int n    = lane & 15;
  int half = lane >> 4;
  bf16x16 b;
#pragma unroll
  for (int e = 0; e < 16; ++e) {
    int k = (e & 7) + 8 * half + ((e >> 3) << 4);
    b[e] = base[n * ld + k];
  }
  return b;
}

// ---------------------------------------------------------------------------
// TDM: 2-D tile load Global -> LDS via tensor descriptor in SGPRs (inline asm
// form per CDNA5 ISA; uniform args land in SGPR tuples).
// ---------------------------------------------------------------------------
static __device__ inline void tdm_load_2d(uint32_t lds_off, const void* gptr,
                                          uint32_t g1_flags,
                                          uint32_t tensor_d0, uint32_t tensor_d1,
                                          uint32_t tile_d0, uint32_t tile_d1,
                                          uint32_t stride0) {
  uint64_t ga = (uint64_t)(uintptr_t)gptr;
  u32x4 g0;
  g0[0] = 1u;                                          // count=1, user D#
  g0[1] = lds_off;                                     // LDS byte address
  g0[2] = (uint32_t)ga;                                // global addr [31:0]
  g0[3] = (uint32_t)((ga >> 32) & 0x01FFFFFFu) | (2u << 30);  // [56:32]|type=2
  u32x8 g1;
  g1[0] = g1_flags;                                    // mask=0|data_size|pad
  g1[1] = (tensor_d0 & 0xFFFFu) << 16;
  g1[2] = ((tensor_d0 >> 16) & 0xFFFFu) | ((tensor_d1 & 0xFFFFu) << 16);
  g1[3] = ((tensor_d1 >> 16) & 0xFFFFu) | ((tile_d0 & 0xFFFFu) << 16);
  g1[4] = (tile_d1 & 0xFFFFu);
  g1[5] = stride0;
  g1[6] = 0u;
  g1[7] = 0u;
  asm volatile("tensor_load_to_lds %0, %1" :: "s"(g0), "s"(g1) : "memory");
}

// data_size=2B, pad_enable, pad_interval=32 dwords (128B row), pad_amount=4
// dwords (16B) -> LDS row stride 144B == [..][72] bf16 rows.
#define TDM_FLAGS_BF16_PAD144 \
  ((1u << 16) | (1u << 20) | (4u << 22) | (3u << 25))

// ---------------------------------------------------------------------------
// Kernel 1: qkv = x @ w_qkv ; scatter into Q/K/V [B,H,N,DH] bf16.
// 256 threads, block tile 128x128, wave tile 32x64, k-step 32, double-buffered.
// ---------------------------------------------------------------------------
__global__ __launch_bounds__(256)
void qkv_gemm_kernel(const float* __restrict__ x, const float* __restrict__ w,
                     __bf16* __restrict__ qb, __bf16* __restrict__ kb,
                     __bf16* __restrict__ vb) {
  __shared__ __bf16 As[2][128][40];    // [buf][row][k], 80B rows
  __shared__ __bf16 Bst[2][128][40];   // [buf][n][k] (transposed), 80B rows

  int tid  = threadIdx.x;
  int wave = tid >> 5;
  int wrow = (wave >> 1) * 32;
  int wcol = (wave & 1) * 64;
  int rowBase = blockIdx.x * 128;      // over B*N = 8192
  int colBase = blockIdx.y * 128;      // over 3072

  // Each thread stages 16 contiguous elements of each tile.
  int ar_r = tid >> 1, ar_c = (tid & 1) * 16;   // A: 128x32
  int br_r = tid >> 3, br_c = (tid & 7) * 16;   // B: 32x128

  float ar[16], br[16];
#pragma unroll
  for (int j = 0; j < 16; ++j)
    ar[j] = x[(size_t)(rowBase + ar_r) * DIM + ar_c + j];
#pragma unroll
  for (int j = 0; j < 16; ++j)
    br[j] = w[(size_t)br_r * QKVC + colBase + br_c + j];
  {
    bf16x8 a0, a1;
#pragma unroll
    for (int j = 0; j < 8; ++j) { a0[j] = (__bf16)ar[j]; a1[j] = (__bf16)ar[8 + j]; }
    *(bf16x8*)&As[0][ar_r][ar_c]     = a0;
    *(bf16x8*)&As[0][ar_r][ar_c + 8] = a1;
#pragma unroll
    for (int j = 0; j < 16; ++j) Bst[0][br_c + j][br_r] = (__bf16)br[j];
  }

  f32x8 acc[2][4] = {};
  int buf = 0;
  for (int k0 = 0; k0 < DIM; k0 += 32) {
    __syncthreads();                       // LDS[buf] ready
    bool hasNext = (k0 + 32) < DIM;
    if (hasNext) {                         // prefetch next tiles into regs
#pragma unroll
      for (int j = 0; j < 16; ++j)
        ar[j] = x[(size_t)(rowBase + ar_r) * DIM + k0 + 32 + ar_c + j];
#pragma unroll
      for (int j = 0; j < 16; ++j)
        br[j] = w[(size_t)(k0 + 32 + br_r) * QKVC + colBase + br_c + j];
    }

    bf16x16 a0 = load_a_frag(&As[buf][wrow][0], 40);
    bf16x16 a1 = load_a_frag(&As[buf][wrow + 16][0], 40);
#pragma unroll
    for (int ni = 0; ni < 4; ++ni) {
      bf16x16 bf = load_bT_frag(&Bst[buf][wcol + ni * 16][0], 40);
      acc[0][ni] = wmma_bf16(a0, bf, acc[0][ni]);
      acc[1][ni] = wmma_bf16(a1, bf, acc[1][ni]);
    }

    if (hasNext) {                         // publish next tiles to LDS[buf^1]
      bf16x8 s0, s1;
#pragma unroll
      for (int j = 0; j < 8; ++j) { s0[j] = (__bf16)ar[j]; s1[j] = (__bf16)ar[8 + j]; }
      *(bf16x8*)&As[buf ^ 1][ar_r][ar_c]     = s0;
      *(bf16x8*)&As[buf ^ 1][ar_r][ar_c + 8] = s1;
#pragma unroll
      for (int j = 0; j < 16; ++j) Bst[buf ^ 1][br_c + j][br_r] = (__bf16)br[j];
    }
    buf ^= 1;
  }

  int lane = tid & 31;
  int half = lane >> 4;
  int cc   = lane & 15;
#pragma unroll
  for (int mi = 0; mi < 2; ++mi)
#pragma unroll
    for (int ni = 0; ni < 4; ++ni)
#pragma unroll
      for (int i = 0; i < 8; ++i) {
        int r = rowBase + wrow + mi * 16 + i + 8 * half;
        int c = colBase + wcol + ni * 16 + cc;
        int which = c >> 10;               // 0=q 1=k 2=v
        int rem   = c & 1023;
        int h = rem >> 6, d = rem & 63;
        int b = r >> 11, n = r & (SEQ - 1);
        __bf16* dst = (which == 0) ? qb : (which == 1) ? kb : vb;
        dst[(((size_t)(b * HEADS + h) * SEQ) + n) * DHEAD + d] =
            (__bf16)acc[mi][ni][i];
      }
}

// ---------------------------------------------------------------------------
// Kernel 2: flash attention, 256 threads / 8 waves, 128-query tile per block.
// K tiles double-buffered through the Tensor Data Mover (async, LDS pad to
// 144B rows); V tiles double-buffered via register prefetch + transposed LDS.
// ---------------------------------------------------------------------------
__global__ __launch_bounds__(256)
void flash_attn_kernel(const __bf16* __restrict__ qb,
                       const __bf16* __restrict__ kb,
                       const __bf16* __restrict__ vb,
                       __bf16* __restrict__ ob) {
  __shared__ __bf16 Ks[2][64][72];     // [buf][key][dh], 144B rows (TDM pad)
  __shared__ __bf16 Vst[2][64][72];    // [buf][dh][key] transposed, padded
  __shared__ __bf16 Ps[8][16][72];     // per-wave P, padded

  int tid  = threadIdx.x;
  int wave = tid >> 5;
  int lane = tid & 31;
  int mrow = lane & 15;
  int half = lane >> 4;

  int qt = blockIdx.x & 15;            // 16 query tiles of 128
  int h  = (blockIdx.x >> 4) & 15;
  int b  = blockIdx.x >> 8;

  size_t head = (size_t)(b * HEADS + h) * SEQ * DHEAD;
  const __bf16* Q = qb + head;
  const __bf16* K = kb + head;
  const __bf16* V = vb + head;

  int qrow0 = qt * 128 + wave * 16;
  uint32_t ksLds[2] = { (uint32_t)(uintptr_t)&Ks[0][0][0],
                        (uint32_t)(uintptr_t)&Ks[1][0][0] };

  // Q in registers, A-fragment layout, DH=64 => two 16x32 fragments.
  bf16x16 qf[2];
#pragma unroll
  for (int c = 0; c < 2; ++c)
#pragma unroll
    for (int e = 0; e < 16; ++e) {
      int kk = (e & 7) + 8 * half + ((e >> 3) << 4) + c * 32;
      qf[c][e] = Q[(qrow0 + mrow) * DHEAD + kk];
    }

  // V staging map: each thread owns 16 contiguous bf16 of the 64x64 tile.
  int vr_r = tid >> 2, vr_c = (tid & 3) * 16;

  // ---- prologue: tile 0 in flight ----
  if (wave == 0)
    tdm_load_2d(ksLds[0], K, TDM_FLAGS_BF16_PAD144,
                DHEAD, SEQ, 64, 64, DHEAD);
  {
    bf16x8 v0 = *(const bf16x8*)&V[(size_t)vr_r * DHEAD + vr_c];
    bf16x8 v1 = *(const bf16x8*)&V[(size_t)vr_r * DHEAD + vr_c + 8];
#pragma unroll
    for (int j = 0; j < 8; ++j) {
      Vst[0][vr_c + j][vr_r]     = v0[j];
      Vst[0][vr_c + 8 + j][vr_r] = v1[j];
    }
  }

  f32x8 o[4] = {};
  float rm[8], rl[8];
#pragma unroll
  for (int i = 0; i < 8; ++i) { rm[i] = -__builtin_inff(); rl[i] = 0.0f; }

  const float scale = 0.125f;  // DH^-0.5
  int buf = 0;

  for (int kb0 = 0; kb0 < SEQ; kb0 += 64) {
    if (wave == 0) __builtin_amdgcn_s_wait_tensorcnt(0);
    __syncthreads();                   // Ks[buf] (TDM) + Vst[buf] published

    bool hasNext = (kb0 + 64) < SEQ;
    bf16x8 v0 = {}, v1 = {};
    if (hasNext) {
      if (wave == 0)                   // async: next K tile while we compute
        tdm_load_2d(ksLds[buf ^ 1], K + (size_t)(kb0 + 64) * DHEAD,
                    TDM_FLAGS_BF16_PAD144, DHEAD, SEQ, 64, 64, DHEAD);
      v0 = *(const bf16x8*)&V[(size_t)(kb0 + 64 + vr_r) * DHEAD + vr_c];
      v1 = *(const bf16x8*)&V[(size_t)(kb0 + 64 + vr_r) * DHEAD + vr_c + 8];
    }

    // S(16x64) = Q * K^T   (logical B(kdim, j) = Ks[buf][j][kdim])
    f32x8 s[4];
#pragma unroll
    for (int t = 0; t < 4; ++t) {
      f32x8 st = {};
#pragma unroll
      for (int c = 0; c < 2; ++c) {
        bf16x16 kf = load_bT_frag(&Ks[buf][t * 16][c * 32], 72);
        st = wmma_bf16(qf[c], kf, st);
      }
      s[t] = st;
    }
#pragma unroll
    for (int t = 0; t < 4; ++t)
#pragma unroll
      for (int i = 0; i < 8; ++i) s[t][i] *= scale;

    // Online softmax; row reductions stay inside 16-lane groups.
#pragma unroll
    for (int i = 0; i < 8; ++i) {
      float mx = s[0][i];
#pragma unroll
      for (int t = 1; t < 4; ++t) mx = fmaxf(mx, s[t][i]);
#pragma unroll
      for (int off = 1; off < 16; off <<= 1)
        mx = fmaxf(mx, __shfl_xor(mx, off, 32));
      float mnew = fmaxf(rm[i], mx);
      float corr = __expf(rm[i] - mnew);
      rm[i] = mnew;
      float sum = 0.0f;
#pragma unroll
      for (int t = 0; t < 4; ++t) {
        float p = __expf(s[t][i] - mnew);
        sum += p;
        Ps[wave][i + 8 * half][t * 16 + mrow] = (__bf16)p;
      }
#pragma unroll
      for (int off = 1; off < 16; off <<= 1)
        sum += __shfl_xor(sum, off, 32);
      rl[i] = rl[i] * corr + sum;
#pragma unroll
      for (int t = 0; t < 4; ++t) o[t][i] *= corr;
    }
    __syncthreads();                   // cross-lane Ps visibility

    // O(16x64) += P(16x64) @ V(64x64)   (B(kk,d) = Vst[buf][d][kk])
    bf16x16 pf[2];
#pragma unroll
    for (int c = 0; c < 2; ++c)
      pf[c] = load_a_frag(&Ps[wave][0][c * 32], 72);
#pragma unroll
    for (int t = 0; t < 4; ++t)
#pragma unroll
      for (int c = 0; c < 2; ++c) {
        bf16x16 vf = load_bT_frag(&Vst[buf][t * 16][c * 32], 72);
        o[t] = wmma_bf16(pf[c], vf, o[t]);
      }

    if (hasNext) {                     // publish next V tile to Vst[buf^1]
#pragma unroll
      for (int j = 0; j < 8; ++j) {
        Vst[buf ^ 1][vr_c + j][vr_r]     = v0[j];
        Vst[buf ^ 1][vr_c + 8 + j][vr_r] = v1[j];
      }
    }
    buf ^= 1;
  }

  // Normalize and store to attn buffer [B, N, H*DH] bf16.
#pragma unroll
  for (int i = 0; i < 8; ++i) {
    float inv = 1.0f / rl[i];
#pragma unroll
    for (int t = 0; t < 4; ++t) o[t][i] *= inv;
  }
#pragma unroll
  for (int t = 0; t < 4; ++t)
#pragma unroll
    for (int i = 0; i < 8; ++i) {
      int n   = qrow0 + i + 8 * half;
      int col = h * DHEAD + t * 16 + mrow;
      ob[((size_t)(b * SEQ + n)) * (HEADS * DHEAD) + col] = (__bf16)o[t][i];
    }
}

// ---------------------------------------------------------------------------
// Kernel 3: out = attn @ w_out + b_out  (attn bf16 [8192,1024], out f32).
// Same double-buffered tiling as kernel 1.
// ---------------------------------------------------------------------------
__global__ __launch_bounds__(256)
void out_gemm_kernel(const __bf16* __restrict__ attn,
                     const float* __restrict__ w,
                     const float* __restrict__ bias,
                     float* __restrict__ out) {
  __shared__ __bf16 As[2][128][40];
  __shared__ __bf16 Bst[2][128][40];

  int tid  = threadIdx.x;
  int wave = tid >> 5;
  int wrow = (wave >> 1) * 32;
  int wcol = (wave & 1) * 64;
  int rowBase = blockIdx.x * 128;      // over 8192
  int colBase = blockIdx.y * 128;      // over 1024

  int ar_r = tid >> 1, ar_c = (tid & 1) * 16;
  int br_r = tid >> 3, br_c = (tid & 7) * 16;

  bf16x8 a0r, a1r;
  float  br_[16];
  a0r = *(const bf16x8*)&attn[(size_t)(rowBase + ar_r) * DIM + ar_c];
  a1r = *(const bf16x8*)&attn[(size_t)(rowBase + ar_r) * DIM + ar_c + 8];
#pragma unroll
  for (int j = 0; j < 16; ++j)
    br_[j] = w[(size_t)br_r * DIM + colBase + br_c + j];
  {
    *(bf16x8*)&As[0][ar_r][ar_c]     = a0r;
    *(bf16x8*)&As[0][ar_r][ar_c + 8] = a1r;
#pragma unroll
    for (int j = 0; j < 16; ++j) Bst[0][br_c + j][br_r] = (__bf16)br_[j];
  }

  f32x8 acc[2][4] = {};
  int buf = 0;
  for (int k0 = 0; k0 < DIM; k0 += 32) {
    __syncthreads();
    bool hasNext = (k0 + 32) < DIM;
    if (hasNext) {
      a0r = *(const bf16x8*)&attn[(size_t)(rowBase + ar_r) * DIM + k0 + 32 + ar_c];
      a1r = *(const bf16x8*)&attn[(size_t)(rowBase + ar_r) * DIM + k0 + 32 + ar_c + 8];
#pragma unroll
      for (int j = 0; j < 16; ++j)
        br_[j] = w[(size_t)(k0 + 32 + br_r) * DIM + colBase + br_c + j];
    }

    bf16x16 a0 = load_a_frag(&As[buf][wrow][0], 40);
    bf16x16 a1 = load_a_frag(&As[buf][wrow + 16][0], 40);
#pragma unroll
    for (int ni = 0; ni < 4; ++ni) {
      bf16x16 bf = load_bT_frag(&Bst[buf][wcol + ni * 16][0], 40);
      acc[0][ni] = wmma_bf16(a0, bf, acc[0][ni]);
      acc[1][ni] = wmma_bf16(a1, bf, acc[1][ni]);
    }

    if (hasNext) {
      *(bf16x8*)&As[buf ^ 1][ar_r][ar_c]     = a0r;
      *(bf16x8*)&As[buf ^ 1][ar_r][ar_c + 8] = a1r;
#pragma unroll
      for (int j = 0; j < 16; ++j) Bst[buf ^ 1][br_c + j][br_r] = (__bf16)br_[j];
    }
    buf ^= 1;
  }

  int lane = tid & 31;
  int half = lane >> 4;
  int cc   = lane & 15;
#pragma unroll
  for (int mi = 0; mi < 2; ++mi)
#pragma unroll
    for (int ni = 0; ni < 4; ++ni)
#pragma unroll
      for (int i = 0; i < 8; ++i) {
        int r = rowBase + wrow + mi * 16 + i + 8 * half;
        int c = colBase + wcol + ni * 16 + cc;
        out[(size_t)r * DIM + c] = acc[mi][ni][i] + bias[c];
      }
}

// ---------------------------------------------------------------------------
extern "C" void kernel_launch(void* const* d_in, const int* in_sizes, int n_in,
                              void* d_out, int out_size, void* d_ws,
                              size_t ws_size, hipStream_t stream) {
  (void)in_sizes; (void)n_in; (void)out_size; (void)ws_size;
  const float* x     = (const float*)d_in[0];
  const float* w_qkv = (const float*)d_in[1];
  const float* w_out = (const float*)d_in[2];
  const float* b_out = (const float*)d_in[3];
  float* out = (float*)d_out;

  // Workspace: Q | K | V (each [B,H,N,DH] bf16 = 16 MB) | attn ([B,N,1024] bf16)
  const size_t HEAD_ELEMS = (size_t)BSZ * HEADS * SEQ * DHEAD;  // 8388608
  __bf16* qb = (__bf16*)d_ws;
  __bf16* kb = qb + HEAD_ELEMS;
  __bf16* vb = kb + HEAD_ELEMS;
  __bf16* ab = vb + HEAD_ELEMS;

  qkv_gemm_kernel<<<dim3((BSZ * SEQ) / 128, QKVC / 128), dim3(256), 0, stream>>>(
      x, w_qkv, qb, kb, vb);
  flash_attn_kernel<<<dim3(BSZ * HEADS * (SEQ / 128)), dim3(256), 0, stream>>>(
      qb, kb, vb, ab);
  out_gemm_kernel<<<dim3((BSZ * SEQ) / 128, DIM / 128), dim3(256), 0, stream>>>(
      ab, w_out, b_out, out);
}
